// RNNModel_89172110999559
// MI455X (gfx1250) — matrix operand order repaired and checked by reading, not compiled, not run on hardware
//
#include <hip/hip_runtime.h>
#include <hip/hip_bf16.h>
#include <math.h>
#include <stdint.h>

#define NTOKEN 33278
#define NINP   400
#define NHID   1150
#define TSEQ   70
#define BATCH  80

typedef unsigned short u16;
typedef __attribute__((ext_vector_type(16))) __bf16 v16bf;
typedef __attribute__((ext_vector_type(8)))  float  v8f;
typedef __attribute__((ext_vector_type(4)))  float  v4f;

// Explicit global (addrspace 1) pointee types: pins codegen to global_load/
// global_store regardless of address-space inference.
typedef __attribute__((address_space(1))) u16         g_u16;
typedef __attribute__((address_space(1))) float       g_f32;
typedef __attribute__((address_space(1))) v4f         g_v4f;

__device__ __forceinline__ u16 f32_to_bf16_rne(float f) {
  unsigned int u = __float_as_uint(f);
  unsigned int r = u + 0x7FFFu + ((u >> 16) & 1u);
  return (u16)(r >> 16);
}

// Prefetch one K-chunk (64B) ahead via the gfx1250 global prefetch op.
__device__ __forceinline__ void gprefetch64(const g_u16* p) {
  asm volatile("global_prefetch_b8 %0, off offset:64"
               :: "v"((unsigned long long)(uintptr_t)p));
}

union FragU {
  v16bf v;
  v4f f[2];
};

// ---------------------------------------------------------------------------
// Pack an fp32 [rows, cols] matrix into zero-padded bf16 [rpad, kpad] (RNE).
// ---------------------------------------------------------------------------
__global__ void pack_bf16_kernel(const float* __restrict__ src, u16* __restrict__ dst,
                                 int rows, int cols, int kpad, long total) {
  long i = (long)blockIdx.x * blockDim.x + threadIdx.x;
  if (i >= total) return;
  int r = (int)(i / kpad);
  int c = (int)(i % kpad);
  float v = (r < rows && c < cols) ? src[(long)r * cols + c] : 0.0f;
  dst[i] = f32_to_bf16_rne(v);
}

// ---------------------------------------------------------------------------
// Embedding gather -> zero-padded bf16 [T*B, kpad]
// ---------------------------------------------------------------------------
__global__ void embed_kernel(const int* __restrict__ tokens, const float* __restrict__ emb,
                             u16* __restrict__ dst, int kpad, long total) {
  long i = (long)blockIdx.x * blockDim.x + threadIdx.x;
  if (i >= total) return;
  int r = (int)(i / kpad);
  int c = (int)(i % kpad);
  float v = 0.0f;
  if (c < NINP) {
    int tok = tokens[r];
    v = emb[(long)tok * NINP + c];
  }
  dst[i] = f32_to_bf16_rne(v);
}

__global__ void zero_u16_kernel(u16* __restrict__ dst, long total) {
  long i = (long)blockIdx.x * blockDim.x + threadIdx.x;
  if (i < total) dst[i] = 0;
}

// ---------------------------------------------------------------------------
// Register-tiled WMMA bf16 GEMM:  C[M, ldc] = A[M, K] @ B[Npad, K]^T (+biases)
//
// Wave tile: (MT*16) x 64  -> MT x 4 accumulators; per 32-K step we load
// MT A-fragments + 4 B-fragments and issue 4*MT WMMAs (0.75 loads/WMMA at
// MT=2). Block = 8 waves, each on its own 64-wide N chunk.
//
// Requirements (guaranteed by padding): K % 32 == 0, Npad % 64 == 0,
// M % (MT*16) == 0. Wave-uniform N guard keeps EXEC all-1s around WMMA.
//
// Fragment layouts per CDNA5 ISA 7.12.2 (bf16 16x16x32):
//   A lane L (row m0 + L&15): 8-elem chunks at k + 8*(L>=16) and +16
//   B lane L (col n0 + L&15): 8-elem chunks at k + 8*(L>=16) and +16
// ---------------------------------------------------------------------------
template <int MT>
__global__ __launch_bounds__(256) void gemm_bf16_wmma(
    const u16* __restrict__ A_, const u16* __restrict__ B_,
    const float* __restrict__ bias_a_, const float* __restrict__ bias_b_,
    float* __restrict__ C_, int Npad, int K, int ldc, int Nvalid) {
  const int wave = threadIdx.x >> 5;
  const int lane = threadIdx.x & 31;
  const int n0 = (blockIdx.x * 8 + wave) * 64;
  if (n0 >= Npad) return;                    // wave-uniform exit, EXEC stays full
  const int m0 = blockIdx.y * (MT * 16);

  const g_u16* A = (const g_u16*)(uintptr_t)A_;
  const g_u16* B = (const g_u16*)(uintptr_t)B_;
  g_f32*       C = (g_f32*)(uintptr_t)C_;
  const g_f32* bias_a = (const g_f32*)(uintptr_t)bias_a_;
  const g_f32* bias_b = (const g_f32*)(uintptr_t)bias_b_;

  const int lhalf = lane >> 4;
  const int lrow = lane & 15;

  const g_u16* pa[MT];
#pragma unroll
  for (int mi = 0; mi < MT; ++mi)
    pa[mi] = A + (size_t)(m0 + mi * 16 + lrow) * K + (lhalf << 3);
  const g_u16* pb[4];
#pragma unroll
  for (int ni = 0; ni < 4; ++ni)
    pb[ni] = B + (size_t)(n0 + ni * 16 + lrow) * K + (lhalf << 3);

  const v8f vzero = {0.f, 0.f, 0.f, 0.f, 0.f, 0.f, 0.f, 0.f};
  v8f acc[MT][4];
#pragma unroll
  for (int mi = 0; mi < MT; ++mi)
#pragma unroll
    for (int ni = 0; ni < 4; ++ni) acc[mi][ni] = vzero;

  for (int k = 0; k < K; k += 32) {
    FragU a[MT], b[4];
#pragma unroll
    for (int mi = 0; mi < MT; ++mi) {
      a[mi].f[0] = *(const g_v4f*)(pa[mi]);       // K = k + 8h + 0..7   -> VGPR 0-3
      a[mi].f[1] = *(const g_v4f*)(pa[mi] + 16);  // K = k + 8h + 16..23 -> VGPR 4-7
      gprefetch64(pa[mi]);                        // global_prefetch_b8, next chunk
      pa[mi] += 32;
    }
#pragma unroll
    for (int ni = 0; ni < 4; ++ni) {
      b[ni].f[0] = *(const g_v4f*)(pb[ni]);       // K = k + 8h + 0..7
      b[ni].f[1] = *(const g_v4f*)(pb[ni] + 16);  // K = k + 8h + 16..23
      gprefetch64(pb[ni]);
      pb[ni] += 32;
    }
#pragma unroll
    for (int mi = 0; mi < MT; ++mi)
#pragma unroll
      for (int ni = 0; ni < 4; ++ni)
        acc[mi][ni] = __builtin_amdgcn_wmma_f32_16x16x32_bf16(
            false, a[mi].v, false, b[ni].v, (short)0, acc[mi][ni], false, false);
  }

  // Epilogue: D layout -> lane covers col n0+ni*16+lrow, rows m0+8*lhalf+v.
#pragma unroll
  for (int ni = 0; ni < 4; ++ni) {
    const int n = n0 + ni * 16 + lrow;
    if (n < Nvalid) {
      float bs = 0.f;
      if (bias_a_) bs += bias_a[n];
      if (bias_b_) bs += bias_b[n];
#pragma unroll
      for (int mi = 0; mi < MT; ++mi) {
        g_f32* cp = C + (size_t)(m0 + mi * 16 + (lhalf << 3)) * ldc + n;
#pragma unroll
        for (int v = 0; v < 8; ++v) cp[(size_t)v * ldc] = acc[mi][ni][v] + bs;
      }
    }
  }
}

// ---------------------------------------------------------------------------
// LSTM pointwise: gates from pre_x[t] + h@Wh^T, update c, emit h (bf16 twice:
// recurrent input buffer + next-layer activation row).
// Gate order (reference): f, i, o, g.
// ---------------------------------------------------------------------------
__global__ void lstm_point_kernel(const float* __restrict__ prex, const float* __restrict__ recp,
                                  float* __restrict__ cstate,
                                  u16* __restrict__ hbf, u16* __restrict__ hout,
                                  int H, int ldp, int kpad, int kpad_out) {
  int i = blockIdx.x * blockDim.x + threadIdx.x;
  if (i >= BATCH * H) return;
  int b = i / H, j = i % H;
  const float* px = prex + (size_t)b * ldp;
  const float* pr = recp + (size_t)b * ldp;
  float f  = 1.f / (1.f + __expf(-(px[j] + pr[j])));
  float in = 1.f / (1.f + __expf(-(px[H + j] + pr[H + j])));
  float o  = 1.f / (1.f + __expf(-(px[2 * H + j] + pr[2 * H + j])));
  float g  = tanhf(px[3 * H + j] + pr[3 * H + j]);
  float c  = f * cstate[i] + in * g;
  float h  = o * tanhf(c);
  cstate[i] = c;
  u16 hb = f32_to_bf16_rne(h);
  hbf[(size_t)b * kpad + j] = hb;
  hout[(size_t)b * kpad_out + j] = hb;
}

// ---------------------------------------------------------------------------
extern "C" void kernel_launch(void* const* d_in, const int* in_sizes, int n_in,
                              void* d_out, int out_size, void* d_ws, size_t ws_size,
                              hipStream_t stream) {
  (void)in_sizes; (void)n_in; (void)out_size; (void)ws_size;

  const int*   tokens = (const int*)d_in[0];
  const float* emb_w  = (const float*)d_in[1];
  const float* dec_b  = (const float*)d_in[2];
  const float* Wi0 = (const float*)d_in[3];
  const float* bi0 = (const float*)d_in[4];
  const float* Wh0 = (const float*)d_in[5];
  const float* bh0 = (const float*)d_in[6];
  const float* Wi1 = (const float*)d_in[7];
  const float* bi1 = (const float*)d_in[8];
  const float* Wh1 = (const float*)d_in[9];
  const float* bh1 = (const float*)d_in[10];
  const float* Wi2 = (const float*)d_in[11];
  const float* bi2 = (const float*)d_in[12];
  const float* Wh2 = (const float*)d_in[13];
  const float* bh2 = (const float*)d_in[14];
  const float* h00 = (const float*)d_in[15];
  const float* c00 = (const float*)d_in[16];
  const float* h01 = (const float*)d_in[17];
  const float* c01 = (const float*)d_in[18];
  const float* h02 = (const float*)d_in[19];
  const float* c02 = (const float*)d_in[20];
  float* out = (float*)d_out;

  const int K400 = 416, K1150 = 1152;   // K padded to 32
  const int N4H  = 4608;                // 4*1150 padded (mult of 64; also pre stride)
  const int NTOKp = 33280;              // 33278 padded (mult of 64)
  const int M = TSEQ * BATCH;           // 5600 (multiple of 32)

  // ---- workspace carve (bump allocator, 256B aligned; ~200 MB total) ----
  char* base = (char*)d_ws;
  size_t off = 0;
  auto wsa = [&](size_t bytes) -> void* {
    void* p = base + off;
    off += (bytes + 255) & ~(size_t)255;
    return p;
  };
  u16*   Wi0p = (u16*)wsa((size_t)N4H * K400 * 2);
  u16*   Wh0p = (u16*)wsa((size_t)N4H * K1150 * 2);
  u16*   Wi1p = (u16*)wsa((size_t)N4H * K1150 * 2);
  u16*   Wh1p = (u16*)wsa((size_t)N4H * K1150 * 2);
  u16*   Wi2p = (u16*)wsa((size_t)1600 * K1150 * 2);
  u16*   Wh2p = (u16*)wsa((size_t)1600 * K400 * 2);
  u16*   embp = (u16*)wsa((size_t)NTOKp * K400 * 2);
  u16*   actA = (u16*)wsa((size_t)M * K1150 * 2);
  u16*   actB = (u16*)wsa((size_t)M * K1150 * 2);
  float* pre  = (float*)wsa((size_t)M * N4H * 4);
  float* recp = (float*)wsa((size_t)BATCH * N4H * 4);
  float* cst  = (float*)wsa((size_t)BATCH * NHID * 4);
  u16*   hbf  = (u16*)wsa((size_t)BATCH * K1150 * 2);

  auto pack = [&](const float* src, u16* dst, int rows, int cols, int rpad, int kpad) {
    long total = (long)rpad * kpad;
    pack_bf16_kernel<<<(unsigned)((total + 255) / 256), 256, 0, stream>>>(
        src, dst, rows, cols, kpad, total);
  };
  // Big GEMMs (M multiple of 32): MT=2 wave tile 32x64.
  auto gemm2 = [&](const u16* A, const u16* B, const float* ba, const float* bb,
                   float* C, int Mrows, int Npad, int K, int ldc, int Nvalid) {
    dim3 grid((unsigned)((Npad / 64 + 7) / 8), (unsigned)(Mrows / 32));
    gemm_bf16_wmma<2><<<grid, 256, 0, stream>>>(A, B, ba, bb, C, Npad, K, ldc, Nvalid);
  };
  // Small-M GEMM (recurrent, M=80): MT=1 wave tile 16x64.
  auto gemm1 = [&](const u16* A, const u16* B, float* C, int Mrows, int Npad, int K,
                   int ldc, int Nvalid) {
    dim3 grid((unsigned)((Npad / 64 + 7) / 8), (unsigned)(Mrows / 16));
    gemm_bf16_wmma<1><<<grid, 256, 0, stream>>>(A, B, nullptr, nullptr, C, Npad, K,
                                                ldc, Nvalid);
  };

  // ---- weight + embedding packing (fp32 -> padded bf16) ----
  pack(Wi0, Wi0p, 4 * NHID, NINP, N4H, K400);
  pack(Wh0, Wh0p, 4 * NHID, NHID, N4H, K1150);
  pack(Wi1, Wi1p, 4 * NHID, NHID, N4H, K1150);
  pack(Wh1, Wh1p, 4 * NHID, NHID, N4H, K1150);
  pack(Wi2, Wi2p, 4 * NINP, NHID, 1600, K1150);
  pack(Wh2, Wh2p, 4 * NINP, NINP, 1600, K400);
  pack(emb_w, embp, NTOKEN, NINP, NTOKp, K400);

  // ---- embedding gather -> actA ----
  {
    long total = (long)M * K400;
    embed_kernel<<<(unsigned)((total + 255) / 256), 256, 0, stream>>>(
        tokens, emb_w, actA, K400, total);
  }

  // ---- one LSTM layer ----
  auto layer = [&](const u16* actIn, int kIn, const u16* Wip, const u16* Whp,
                   const float* bi, const float* bh, int H, int kH, int Npad,
                   const float* h0, const float* c0, u16* actOut, int kOut) {
    const int Nv = 4 * H;
    // batched input projection: pre = actIn @ Wi^T + bi + bh
    gemm2(actIn, Wip, bi, bh, pre, M, Npad, kIn, N4H, Nv);
    // state init
    hipMemcpyAsync(cst, c0, (size_t)BATCH * H * 4, hipMemcpyDeviceToDevice, stream);
    pack(h0, hbf, BATCH, H, BATCH, kH);
    {
      long total = (long)M * kOut;
      zero_u16_kernel<<<(unsigned)((total + 255) / 256), 256, 0, stream>>>(actOut, total);
    }
    const int pts = BATCH * H;
    for (int t = 0; t < TSEQ; ++t) {
      gemm1(hbf, Whp, recp, BATCH, Npad, kH, N4H, Nv);
      lstm_point_kernel<<<(pts + 255) / 256, 256, 0, stream>>>(
          pre + (size_t)t * BATCH * N4H, recp, cst, hbf,
          actOut + (size_t)t * BATCH * kOut, H, N4H, kH, kOut);
    }
  };

  layer(actA, K400,  Wi0p, Wh0p, bi0, bh0, NHID, K1150, N4H,  h00, c00, actB, K1150);
  layer(actB, K1150, Wi1p, Wh1p, bi1, bh1, NHID, K1150, N4H,  h01, c01, actA, K1150);
  layer(actA, K1150, Wi2p, Wh2p, bi2, bh2, NINP, K400,  1600, h02, c02, actB, K400);

  // ---- tied decoder: out = h3 @ emb_w^T + dec_b  (745 MB fp32 write) ----
  gemm2(actB, embp, dec_b, nullptr, out, M, NTOKp, K400, NTOKEN, NTOKEN);
}
